// PointTransformerLayer_34926674051780
// MI455X (gfx1250) — matrix operand order repaired and checked by reading, not compiled, not run on hardware
//
#include <hip/hip_runtime.h>
#include <hip/hip_bf16.h>

// Problem constants (B, N, C, K) from the reference setup_inputs().
#define BSZ  4
#define NPTS 4096
#define CDIM 128
#define KNN  16

typedef __attribute__((ext_vector_type(16))) _Float16 v16h;
typedef __attribute__((ext_vector_type(8)))  _Float16 v8h;
typedef __attribute__((ext_vector_type(8)))  float    v8f;

// Workspace layout: [ knn indices: B*N*K ints = 1 MB | f16 weights: 96 KB ]
constexpr size_t WS_KNN_BYTES = (size_t)BSZ * NPTS * KNN * 4;
constexpr int    WMAT         = CDIM * CDIM;  // 16384 elements per matrix

// ------------------------------------------------------------------
// Kernel 0: one-time f32 -> f16 weight conversion (phi, g1, g2) so the
// fused kernel stages pure f16 via b128 copies (half the L2 traffic,
// zero per-WG conversion VALU).
// ------------------------------------------------------------------
__global__ void prep_weights_kernel(const float* __restrict__ phi_w,
                                    const float* __restrict__ g1_w,
                                    const float* __restrict__ g2_w,
                                    _Float16* __restrict__ wf16) {
  const int i = blockIdx.x * 256 + threadIdx.x;  // [0, 16384)
  wf16[i]            = (_Float16)phi_w[i];
  wf16[WMAT + i]     = (_Float16)g1_w[i];
  wf16[2 * WMAT + i] = (_Float16)g2_w[i];
}

// ------------------------------------------------------------------
// Kernel 1: KNN (k=16 incl. self). 1 thread per query point, coords
// tiled through LDS, branch-free insertion into sorted top-16.
// ------------------------------------------------------------------
__global__ void knn_kernel(const float* __restrict__ coords,
                           int* __restrict__ knn_out) {
  __shared__ float tile[128][3];
  const int b = blockIdx.y;
  const int q = blockIdx.x * 128 + threadIdx.x;
  const float* cb = coords + (size_t)b * NPTS * 3;
  const float qx = cb[q * 3 + 0], qy = cb[q * 3 + 1], qz = cb[q * 3 + 2];

  float bd[KNN];
  int   bi[KNN];
#pragma unroll
  for (int s = 0; s < KNN; ++s) { bd[s] = 3.4e38f; bi[s] = 0; }

  for (int t0 = 0; t0 < NPTS; t0 += 128) {
    __syncthreads();
    tile[threadIdx.x][0] = cb[(t0 + threadIdx.x) * 3 + 0];
    tile[threadIdx.x][1] = cb[(t0 + threadIdx.x) * 3 + 1];
    tile[threadIdx.x][2] = cb[(t0 + threadIdx.x) * 3 + 2];
    __syncthreads();
    for (int tt = 0; tt < 128; ++tt) {
      const float dx = qx - tile[tt][0];
      const float dy = qy - tile[tt][1];
      const float dz = qz - tile[tt][2];
      const float d  = dx * dx + dy * dy + dz * dz;
      const int  ci  = t0 + tt;
      if (d < bd[KNN - 1]) {
        // branch-free sorted insert (bd ascending); reads bd[s-1] before
        // it is overwritten since we iterate s downward.
#pragma unroll
        for (int s = KNN - 1; s >= 0; --s) {
          const bool lt    = d < bd[s];
          const bool place = (s == 0) || (d >= bd[s - 1]);
          if (lt) {
            if (place) { bd[s] = d;         bi[s] = ci; }
            else       { bd[s] = bd[s - 1]; bi[s] = bi[s - 1]; }
          }
        }
      }
    }
  }
  int* o = knn_out + ((size_t)b * NPTS + q) * KNN;
#pragma unroll
  for (int s = 0; s < KNN; ++s) o[s] = bi[s];
}

// ------------------------------------------------------------------
// LDS layout for the fused kernel (dynamic shared). 32 points / WG.
// ------------------------------------------------------------------
constexpr int OFF_W    = 0;                          // phi|g1|g2 f16 = 96 KB
constexpr int OFF_WPHI = OFF_W;
constexpr int OFF_WG1  = OFF_W + WMAT * 2;
constexpr int OFF_WG2  = OFF_W + 2 * WMAT * 2;
constexpr int OFF_ATT  = OFF_W + 3 * WMAT * 2;       // 512x128 f16 = 128 KB
constexpr int OFF_XI   = OFF_ATT + 512 * CDIM * 2;   // 32x128 f32 = 16 KB
constexpr int OFF_FEAT = OFF_XI + 32 * CDIM * 4;     // 32x128 f16 = 8 KB
constexpr int OFF_GEO  = OFF_FEAT + 32 * CDIM * 2;   // 512x8 f32 = 16 KB
constexpr int OFF_SM   = OFF_GEO + 512 * 8 * 4;      // small params (f32)
constexpr int NSMALL   = 1952;                       // floats
constexpr int SMEM_BYTES = OFF_SM + NSMALL * 4;      // ~271.6 KB (< 320 KB/WGP)

// small-parameter float offsets inside sS
constexpr int S_S1W = 0;     // 9
constexpr int S_S1B = 12;    // 3
constexpr int S_S2W = 16;    // 384
constexpr int S_S2B = 400;   // 128
constexpr int S_PSW = 528;   // 384
constexpr int S_PSB = 912;   // 128
constexpr int S_ALW = 1040;  // 384
constexpr int S_ALB = 1424;  // 128
constexpr int S_PHB = 1552;  // 128
constexpr int S_G1B = 1680;  // 128
constexpr int S_G2B = 1808;  // 128

// ---- WMMA fragment loaders (contiguous-k order k = 16h+s, f16 16x16x32) ----
// A (16x32): lane m=lane&15 / h=lane>>4 holds K = h*16..h*16+15
// of row m (matches load_B) -> two 16B LDS chunks.
__device__ inline v16h load_A(const _Float16* base /*row0, stride CDIM*/,
                              int kbase, int lane) {
  const int m = lane & 15, h = lane >> 4;
  const _Float16* r = base + m * CDIM + kbase + h * 16;
  v8h lo = *(const v8h*)(r);
  v8h hi = *(const v8h*)(r + 8);
  v16h a;
#pragma unroll
  for (int i = 0; i < 8; ++i) { a[i] = lo[i]; a[i + 8] = hi[i]; }
  return a;
}
// B (32x16): B[k][n] = w[n][k] (torch Linear (out,in) row-major). Lane
// n=lane&15 / kh=lane>>4 holds K = kh*16..kh*16+15 of column n, which is
// 16 contiguous f16 of w's row n -> one 32B LDS read.
__device__ inline v16h load_B(const _Float16* w, int ntile, int kbase, int lane) {
  const int n = lane & 15, kh = lane >> 4;
  return *(const v16h*)(w + (size_t)(ntile * 16 + n) * CDIM + kbase + kh * 16);
}

__device__ inline v8f wmma_f16(v16h a, v16h b, v8f c) {
  return __builtin_amdgcn_wmma_f32_16x16x32_f16(false, a, false, b,
                                                (short)0, c, false, false);
}
__device__ inline v8f zero8() {
  v8f z;
#pragma unroll
  for (int i = 0; i < 8; ++i) z[i] = 0.0f;
  return z;
}

// ------------------------------------------------------------------
// Kernel 2: fused point-transformer block. One WG = 32 query points.
// 512 threads = 16 wave32; wave w owns att rows [32w, 32w+32) = the 16
// neighbors of points 2w and 2w+1 (so softmax + j-sum are wave-local).
// ------------------------------------------------------------------
__global__ void __launch_bounds__(512)
pt_fused_kernel(const float* __restrict__ coords,
                const float* __restrict__ features,
                const _Float16* __restrict__ wf16,   // [phi|g1|g2] f16
                const float* __restrict__ phi_b,
                const float* __restrict__ psi_w, const float* __restrict__ psi_b,
                const float* __restrict__ g1_b,  const float* __restrict__ g2_b,
                const float* __restrict__ s1_w,  const float* __restrict__ s1_b,
                const float* __restrict__ s2_w,  const float* __restrict__ s2_b,
                const float* __restrict__ al_w,  const float* __restrict__ al_b,
                const int* __restrict__ knn, float* __restrict__ out) {
  extern __shared__ char smem[];
  _Float16* sWphi = (_Float16*)(smem + OFF_WPHI);
  _Float16* sWg1  = (_Float16*)(smem + OFF_WG1);
  _Float16* sWg2  = (_Float16*)(smem + OFF_WG2);
  _Float16* sAtt  = (_Float16*)(smem + OFF_ATT);
  float*    sXi   = (float*)(smem + OFF_XI);
  _Float16* sFeat = (_Float16*)(smem + OFF_FEAT);
  float*    sGeo  = (float*)(smem + OFF_GEO);
  float*    sS    = (float*)(smem + OFF_SM);

  const int tid = threadIdx.x;
  const int b   = blockIdx.y;
  const int n0  = blockIdx.x * 32;

  // ---- Phase 0: stage pre-converted f16 weights via b128 copies ----
  {
    const uint4* gw = (const uint4*)wf16;          // 3*16384 halves
    uint4*       sw = (uint4*)(smem + OFF_W);
#pragma unroll
    for (int k = 0; k < 12; ++k)                   // 6144 uint4 / 512 thr
      sw[k * 512 + tid] = gw[k * 512 + tid];
  }
  for (int i = tid; i < 32 * CDIM; i += 512)
    sFeat[i] = (_Float16)features[((size_t)b * NPTS + n0) * CDIM + i];
  if (tid < 9) sS[S_S1W + tid] = s1_w[tid];
  if (tid < 3) sS[S_S1B + tid] = s1_b[tid];
  for (int i = tid; i < 384; i += 512) {
    sS[S_S2W + i] = s2_w[i]; sS[S_PSW + i] = psi_w[i]; sS[S_ALW + i] = al_w[i];
  }
  for (int i = tid; i < 128; i += 512) {
    sS[S_S2B + i] = s2_b[i]; sS[S_PSB + i] = psi_b[i]; sS[S_ALB + i] = al_b[i];
    sS[S_PHB + i] = phi_b[i]; sS[S_G1B + i] = g1_b[i]; sS[S_G2B + i] = g2_b[i];
  }
  __syncthreads();

  const int wv = tid >> 5, lane = tid & 31;
  const int hh = lane >> 4, nn = lane & 15;

  // ---- Phase 1a: xi = phi(features): 32x128 @ 128x128.
  // 16 waves -> 16 output tiles: wave w does M-tile w>>3, N-tile w&7.
  {
    const int mtx = wv >> 3, ntx = wv & 7;
    v8f acc = zero8();
#pragma unroll
    for (int ks = 0; ks < 4; ++ks) {
      v16h a  = load_A(sFeat + (size_t)(mtx * 16) * CDIM, ks * 32, lane);
      v16h bf = load_B(sWphi, ntx, ks * 32, lane);
      acc = wmma_f16(a, bf, acc);
    }
#pragma unroll
    for (int v = 0; v < 8; ++v)  // C/D layout: m = v + 8*hh, n = nn
      sXi[(mtx * 16 + v + 8 * hh) * CDIM + ntx * 16 + nn] =
          acc[v] + sS[S_PHB + ntx * 16 + nn];
  }
  __syncthreads();

  // ---- Phase 1b: per (point,neighbor) row: build att = xi - psi + pos ----
  {
    const int r = tid, p = r >> 4, j = r & 15;
    const int ng = n0 + p;
    const int nb = knn[((size_t)b * NPTS + ng) * KNN + j];
    const float* cb = coords + (size_t)b * NPTS * 3;
    const float qx = cb[ng * 3 + 0], qy = cb[ng * 3 + 1], qz = cb[ng * 3 + 2];
    const float cx = cb[nb * 3 + 0], cy = cb[nb * 3 + 1], cz = cb[nb * 3 + 2];
    const float rx = qx - cx, ry = qy - cy, rz = qz - cz;
    const float h0 = fmaxf(0.f, sS[S_S1W + 0] * rx + sS[S_S1W + 1] * ry + sS[S_S1W + 2] * rz + sS[S_S1B + 0]);
    const float h1 = fmaxf(0.f, sS[S_S1W + 3] * rx + sS[S_S1W + 4] * ry + sS[S_S1W + 5] * rz + sS[S_S1B + 1]);
    const float h2 = fmaxf(0.f, sS[S_S1W + 6] * rx + sS[S_S1W + 7] * ry + sS[S_S1W + 8] * rz + sS[S_S1B + 2]);
    sGeo[r * 8 + 0] = cx; sGeo[r * 8 + 1] = cy; sGeo[r * 8 + 2] = cz;
    sGeo[r * 8 + 3] = h0; sGeo[r * 8 + 4] = h1; sGeo[r * 8 + 5] = h2;
    for (int c = 0; c < CDIM; ++c) {
      const float pos = sS[S_S2W + c * 3 + 0] * h0 + sS[S_S2W + c * 3 + 1] * h1 +
                        sS[S_S2W + c * 3 + 2] * h2 + sS[S_S2B + c];
      const float xj  = sS[S_PSW + c * 3 + 0] * cx + sS[S_PSW + c * 3 + 1] * cy +
                        sS[S_PSW + c * 3 + 2] * cz + sS[S_PSB + c];
      sAtt[r * CDIM + c] = (_Float16)(sXi[p * CDIM + c] - xj + pos);
    }
  }
  __syncthreads();

  const _Float16* Arow0 = sAtt + (size_t)(wv * 32) * CDIM;
  const _Float16* Arow1 = sAtt + (size_t)(wv * 32 + 16) * CDIM;
  v8f acc[2][8];

  // ---- Phase 2a: y1 = relu(att @ g1_w^T + g1_b), in-place (own rows) ----
#pragma unroll
  for (int mt = 0; mt < 2; ++mt)
#pragma unroll
    for (int nt = 0; nt < 8; ++nt) acc[mt][nt] = zero8();
#pragma unroll
  for (int ks = 0; ks < 4; ++ks) {
    v16h a0 = load_A(Arow0, ks * 32, lane);
    v16h a1 = load_A(Arow1, ks * 32, lane);
#pragma unroll
    for (int nt = 0; nt < 8; ++nt) {
      v16h bf = load_B(sWg1, nt, ks * 32, lane);
      acc[0][nt] = wmma_f16(a0, bf, acc[0][nt]);
      acc[1][nt] = wmma_f16(a1, bf, acc[1][nt]);
    }
  }
#pragma unroll
  for (int mt = 0; mt < 2; ++mt)
#pragma unroll
    for (int nt = 0; nt < 8; ++nt) {
      const float bia = sS[S_G1B + nt * 16 + nn];
#pragma unroll
      for (int v = 0; v < 8; ++v) {
        const float y = fmaxf(acc[mt][nt][v] + bia, 0.f);
        sAtt[(size_t)(wv * 32 + mt * 16 + v + 8 * hh) * CDIM + nt * 16 + nn] = (_Float16)y;
      }
    }
  // wave-local LDS RAW: DS ops are in-order within a wave, no barrier needed.

  // ---- Phase 2b: z = y1 @ g2_w^T + g2_b (kept in C/D registers) ----
#pragma unroll
  for (int mt = 0; mt < 2; ++mt)
#pragma unroll
    for (int nt = 0; nt < 8; ++nt) acc[mt][nt] = zero8();
#pragma unroll
  for (int ks = 0; ks < 4; ++ks) {
    v16h a0 = load_A(Arow0, ks * 32, lane);
    v16h a1 = load_A(Arow1, ks * 32, lane);
#pragma unroll
    for (int nt = 0; nt < 8; ++nt) {
      v16h bf = load_B(sWg2, nt, ks * 32, lane);
      acc[0][nt] = wmma_f16(a0, bf, acc[0][nt]);
      acc[1][nt] = wmma_f16(a1, bf, acc[1][nt]);
    }
  }
#pragma unroll
  for (int mt = 0; mt < 2; ++mt)
#pragma unroll
    for (int nt = 0; nt < 8; ++nt) {
      const float bia = sS[S_G2B + nt * 16 + nn];
#pragma unroll
      for (int v = 0; v < 8; ++v) acc[mt][nt][v] += bia;
    }

  // ---- Phase 3: channel softmax + sum_j att*val, all wave-local ----
#pragma unroll
  for (int mt = 0; mt < 2; ++mt) {
    // softmax over 128 channels of each row m = v + 8*hh: reduce over the
    // 8 N-tile regs then across the 16 lanes sharing hh (xor 1,2,4,8).
#pragma unroll
    for (int v = 0; v < 8; ++v) {
      float mx = -3.4e38f;
#pragma unroll
      for (int nt = 0; nt < 8; ++nt) mx = fmaxf(mx, acc[mt][nt][v]);
#pragma unroll
      for (int s = 1; s < 16; s <<= 1) mx = fmaxf(mx, __shfl_xor(mx, s, 32));
      float sum = 0.f;
#pragma unroll
      for (int nt = 0; nt < 8; ++nt) {
        const float e = __expf(acc[mt][nt][v] - mx);
        acc[mt][nt][v] = e;
        sum += e;
      }
#pragma unroll
      for (int s = 1; s < 16; s <<= 1) sum += __shfl_xor(sum, s, 32);
      const float inv = 1.f / sum;
#pragma unroll
      for (int nt = 0; nt < 8; ++nt) acc[mt][nt][v] *= inv;
    }
    // out[p][c] = sum over the 16 neighbor rows of att*val; val recomputed
    // from 6 cached floats/row (alpha(c_n)+bias+pos).
    float outp[8];
#pragma unroll
    for (int nt = 0; nt < 8; ++nt) outp[nt] = 0.f;
#pragma unroll
    for (int v = 0; v < 8; ++v) {
      const int row = wv * 32 + mt * 16 + v + 8 * hh;
      const float cx = sGeo[row * 8 + 0], cy = sGeo[row * 8 + 1], cz = sGeo[row * 8 + 2];
      const float h0 = sGeo[row * 8 + 3], h1 = sGeo[row * 8 + 4], h2 = sGeo[row * 8 + 5];
#pragma unroll
      for (int nt = 0; nt < 8; ++nt) {
        const int c = nt * 16 + nn;
        const float pos = sS[S_S2W + c * 3 + 0] * h0 + sS[S_S2W + c * 3 + 1] * h1 +
                          sS[S_S2W + c * 3 + 2] * h2 + sS[S_S2B + c];
        const float val = sS[S_ALW + c * 3 + 0] * cx + sS[S_ALW + c * 3 + 1] * cy +
                          sS[S_ALW + c * 3 + 2] * cz + sS[S_ALB + c] + pos;
        outp[nt] += acc[mt][nt][v] * val;
      }
    }
    const int pt = n0 + wv * 2 + mt;
    float* orow = out + ((size_t)b * NPTS + pt) * CDIM;
#pragma unroll
    for (int nt = 0; nt < 8; ++nt) {
      const float t = outp[nt] + __shfl_xor(outp[nt], 16, 32); // merge m-halves
      if (lane < 16) orow[nt * 16 + nn] = t;
    }
  }
}

// ------------------------------------------------------------------
extern "C" void kernel_launch(void* const* d_in, const int* in_sizes, int n_in,
                              void* d_out, int out_size, void* d_ws, size_t ws_size,
                              hipStream_t stream) {
  const float* coords   = (const float*)d_in[0];
  const float* features = (const float*)d_in[1];
  const float* phi_w    = (const float*)d_in[2];
  const float* phi_b    = (const float*)d_in[3];
  const float* psi_w    = (const float*)d_in[4];
  const float* psi_b    = (const float*)d_in[5];
  const float* g1_w     = (const float*)d_in[6];
  const float* g1_b     = (const float*)d_in[7];
  const float* g2_w     = (const float*)d_in[8];
  const float* g2_b     = (const float*)d_in[9];
  const float* s1_w     = (const float*)d_in[10];
  const float* s1_b     = (const float*)d_in[11];
  const float* s2_w     = (const float*)d_in[12];
  const float* s2_b     = (const float*)d_in[13];
  const float* al_w     = (const float*)d_in[14];
  const float* al_b     = (const float*)d_in[15];

  int*      knn  = (int*)d_ws;                              // 1 MB
  _Float16* wf16 = (_Float16*)((char*)d_ws + WS_KNN_BYTES); // 96 KB

  prep_weights_kernel<<<WMAT / 256, 256, 0, stream>>>(phi_w, g1_w, g2_w, wf16);

  dim3 gk(NPTS / 128, BSZ);
  knn_kernel<<<gk, 128, 0, stream>>>(coords, knn);

  dim3 gp(NPTS / 32, BSZ);
  pt_fused_kernel<<<gp, 512, SMEM_BYTES, stream>>>(
      coords, features, wf16, phi_b, psi_w, psi_b, g1_b, g2_b,
      s1_w, s1_b, s2_w, s2_b, al_w, al_b, knn, (float*)d_out);
}
